// GraphConvolution_90589450207873
// MI455X (gfx1250) — matrix operand order repaired and checked by reading, not compiled, not run on hardware
//
#include <hip/hip_runtime.h>

// ---------------------------------------------------------------------------
// Fused temporal-GCN layer for MI455X (gfx1250, wave32, WMMA).
// out[b,t,n,u] = sum_s ( A[s,t] @ X[b,t] )[n,:] @ K_s   (f16 WMMA, f32 accum)
// ---------------------------------------------------------------------------

typedef _Float16 v16h __attribute__((ext_vector_type(16)));
typedef float    v8f  __attribute__((ext_vector_type(8)));
typedef _Float16 h2   __attribute__((ext_vector_type(2)));
typedef _Float16 h4   __attribute__((ext_vector_type(4)));
typedef float    f2   __attribute__((ext_vector_type(2)));
typedef float    f4   __attribute__((ext_vector_type(4)));

namespace gcn {
constexpr int Bc = 8, Tc = 32, Nc = 512, Dc = 64, Sc = 3, Uc = 64;
constexpr int MT = 128;          // output-row tile per workgroup
constexpr int KT = 32;           // WMMA K-chunk (16x16x32 f16)
constexpr int LDA = 36;          // lA row stride (halves): 72B rows (8B-mult), conflict-free
constexpr int LDX = 36;          // lX row stride
constexpr int LDP = 66;          // lP row stride: 132B rows, gcd(33,64)=1
constexpr int LDK = 194;         // lK row stride: 388B rows, gcd(97,64)=1
}

// A-operand fragment: 16x32 f16, row M = lane%16 lives in one lane.
// VGPR v (pair 2v,2v+1):  K = 16*(v/4) + 8*(lane/16) + 2*(v&3) (+1)
static __device__ inline v16h frag_row_A(const _Float16* __restrict__ p, int h) {
  v16h a;
#pragma unroll
  for (int v = 0; v < 8; ++v) {
    const int k = ((v & 4) << 2) + (h << 3) + ((v & 3) << 1);
    h2 pr = *(const h2*)(p + k);
    a[2 * v] = pr.x; a[2 * v + 1] = pr.y;
  }
  return a;
}

// B-operand fragment: 32x16 f16, column N = lane%16.
// VGPR v (pair 2v,2v+1):  K = 16*(v/4) + 8*(lane/16) + 2*(v&3) (+1)
static __device__ inline v16h frag_col_B(const _Float16* __restrict__ p, int h) {
  v16h b;
#pragma unroll
  for (int v = 0; v < 8; ++v) {
    const int k = ((v & 4) << 2) + (h << 3) + ((v & 3) << 1);
    h2 pr = *(const h2*)(p + k);
    b[2 * v] = pr.x; b[2 * v + 1] = pr.y;
  }
  return b;
}

__global__ __launch_bounds__(256)
void gcn_fused_kernel(const float* __restrict__ feat,   // [B,T,N,D]
                      const float* __restrict__ adj,    // [S,T,N,N]
                      const float* __restrict__ kern,   // [S*D, U]
                      float* __restrict__ out)          // [B,T,N,U]
{
  using namespace gcn;
  __shared__ _Float16 lA[MT * LDA];   // A tile  [128][32]   (rows n, cols m)
  __shared__ _Float16 lX[Dc * LDX];   // X^T tile [64][32]   (rows d, cols m)
  __shared__ _Float16 lP[MT * LDP];   // P_s tile [128][64]  (rows n, cols d)
  __shared__ _Float16 lK[Uc * LDK];   // kernel^T [64][192]  (rows u, cols f)

  const int tid  = threadIdx.x;
  const int lane = tid & 31;
  const int wave = tid >> 5;
  const int ln   = lane & 15;        // position within half-wave
  const int h    = lane >> 4;        // half-wave index
  const int wm   = wave & 3;         // wave's 32-row block within 128
  const int wn   = wave >> 2;        // wave's 32-col block within 64

  const int mt = blockIdx.x & 3;     // M-tile within N=512
  const int bt = blockIdx.x >> 2;    // flattened (b*T + t), 0..255
  const int t  = bt & (Tc - 1);
  const int m0 = mt * MT;

  // Staging decompositions (constant per thread)
  const int aj4 = tid & 7;           // A: 4-column group (16B global / 8B LDS)
  const int ai0 = tid >> 3;          // A: row 0..31 (+32r)
  const int xd  = (tid & 31) * 2;    // X: even d
  const int xjp = (tid >> 5) * 2;    // X: even j base (+16r)

  // Preload projection kernel, transposed: lK[u][f] = kern[f][u]
  for (int idx = tid; idx < Sc * Dc * Uc; idx += 256) {
    const int f = idx >> 6;          // 0..191
    const int u = idx & 63;
    lK[u * LDK + f] = (_Float16)kern[idx];
  }

  v8f Oacc[2][2];
#pragma unroll
  for (int fi = 0; fi < 2; ++fi)
#pragma unroll
    for (int fj = 0; fj < 2; ++fj)
      Oacc[fi][fj] = (v8f){0.f, 0.f, 0.f, 0.f, 0.f, 0.f, 0.f, 0.f};

#pragma unroll 1
  for (int s = 0; s < Sc; ++s) {
    // ---- Stage 1: P_s = A[s,t, m0:m0+128, :] @ X[b,t]  (128x64 f32) ----
    v8f P[2][2];
#pragma unroll
    for (int fi = 0; fi < 2; ++fi)
#pragma unroll
      for (int fj = 0; fj < 2; ++fj)
        P[fi][fj] = (v8f){0.f, 0.f, 0.f, 0.f, 0.f, 0.f, 0.f, 0.f};

#pragma unroll 1
    for (int kc = 0; kc < Nc / KT; ++kc) {
      __syncthreads();               // protect lA/lX from previous readers
      {  // A tile: 128x32.  b128 loads (128B/8 threads), h4 pack, ds_store_b64
        const size_t rowbase =
            ((size_t)(s * Tc + t) * Nc + (size_t)(m0 + ai0)) * Nc +
            kc * KT + 4 * aj4;
#pragma unroll
        for (int r = 0; r < 4; ++r) {
          f4 a = *(const f4*)(adj + rowbase + (size_t)r * 32 * Nc);
          h4 q;
          q.x = (_Float16)a.x; q.y = (_Float16)a.y;
          q.z = (_Float16)a.z; q.w = (_Float16)a.w;
          *(h4*)&lA[(ai0 + r * 32) * LDA + 4 * aj4] = q;
        }
      }
      {  // X tile: 32x64 -> lX[d][j] transposed.  b64 loads, packed b32 stores
#pragma unroll
        for (int r = 0; r < 2; ++r) {
          const int j = xjp + 16 * r;
          const size_t base = ((size_t)bt * Nc + kc * KT + j) * Dc + xd;
          f2 a0 = *(const f2*)(feat + base);        // row j,   d..d+1
          f2 a1 = *(const f2*)(feat + base + Dc);   // row j+1, d..d+1
          h2 p0; p0.x = (_Float16)a0.x; p0.y = (_Float16)a1.x;
          h2 p1; p1.x = (_Float16)a0.y; p1.y = (_Float16)a1.y;
          *(h2*)&lX[xd * LDX + j]       = p0;
          *(h2*)&lX[(xd + 1) * LDX + j] = p1;
        }
      }
      __syncthreads();

      v16h af[2], bf[2];
#pragma unroll
      for (int fi = 0; fi < 2; ++fi)
        af[fi] = frag_row_A(lA + (wm * 32 + fi * 16 + ln) * LDA, h);
#pragma unroll
      for (int fj = 0; fj < 2; ++fj)
        bf[fj] = frag_col_B(lX + (wn * 32 + fj * 16 + ln) * LDX, h);

#pragma unroll
      for (int fi = 0; fi < 2; ++fi)
#pragma unroll
        for (int fj = 0; fj < 2; ++fj)
          P[fi][fj] = __builtin_amdgcn_wmma_f32_16x16x32_f16(
              false, af[fi], false, bf[fj], (short)0, P[fi][fj], false, false);
    }

    // ---- Spill P_s to LDS as f16 (C-layout -> row-major) ----
    __syncthreads();
#pragma unroll
    for (int fi = 0; fi < 2; ++fi)
#pragma unroll
      for (int fj = 0; fj < 2; ++fj)
#pragma unroll
        for (int r = 0; r < 8; ++r)
          lP[(wm * 32 + fi * 16 + 8 * h + r) * LDP + wn * 32 + fj * 16 + ln] =
              (_Float16)P[fi][fj][r];
    __syncthreads();

    // ---- Stage 2: Oacc += P_s (128x64) @ K_s (64x64), K-chunks of 32 ----
#pragma unroll
    for (int kc2 = 0; kc2 < Dc / KT; ++kc2) {
      v16h a2[2], b2[2];
#pragma unroll
      for (int fi = 0; fi < 2; ++fi)
        a2[fi] = frag_row_A(lP + (wm * 32 + fi * 16 + ln) * LDP + kc2 * KT, h);
#pragma unroll
      for (int fj = 0; fj < 2; ++fj)
        b2[fj] = frag_col_B(lK + (wn * 32 + fj * 16 + ln) * LDK + s * Dc + kc2 * KT, h);

#pragma unroll
      for (int fi = 0; fi < 2; ++fi)
#pragma unroll
        for (int fj = 0; fj < 2; ++fj)
          Oacc[fi][fj] = __builtin_amdgcn_wmma_f32_16x16x32_f16(
              false, a2[fi], false, b2[fj], (short)0, Oacc[fi][fj], false, false);
    }
  }

  // ---- Write output tile (f32) ----
#pragma unroll
  for (int fi = 0; fi < 2; ++fi)
#pragma unroll
    for (int fj = 0; fj < 2; ++fj)
#pragma unroll
      for (int r = 0; r < 8; ++r) {
        const int mg = m0 + wm * 32 + fi * 16 + 8 * h + r;
        const int u  = wn * 32 + fj * 16 + ln;
        out[((size_t)bt * gcn::Nc + mg) * gcn::Uc + u] = Oacc[fi][fj][r];
      }
}

extern "C" void kernel_launch(void* const* d_in, const int* in_sizes, int n_in,
                              void* d_out, int out_size, void* d_ws, size_t ws_size,
                              hipStream_t stream) {
  (void)in_sizes; (void)n_in; (void)d_ws; (void)ws_size; (void)out_size;
  const float* feat = (const float*)d_in[0];   // [8,32,512,64]
  const float* adj  = (const float*)d_in[1];   // [3,32,512,512]
  const float* kern = (const float*)d_in[2];   // [192,64]
  float* out = (float*)d_out;                  // [8,32,512,64]

  const int grid = gcn::Bc * gcn::Tc * (gcn::Nc / gcn::MT);  // 1024 workgroups
  gcn_fused_kernel<<<grid, 256, 0, stream>>>(feat, adj, kern, out);
}
